// Encoder_3075196584282
// MI455X (gfx1250) — matrix operand order repaired and checked by reading, not compiled, not run on hardware
//
#include <hip/hip_runtime.h>
#include <math.h>

// ---------- types ----------
typedef __attribute__((ext_vector_type(16))) __bf16 v16bf;
typedef __attribute__((ext_vector_type(8)))  float  v8f;
typedef __attribute__((ext_vector_type(4)))  unsigned int v4u;
typedef __attribute__((ext_vector_type(4)))  unsigned int u32x4;
typedef __attribute__((ext_vector_type(8)))  int i32x8;
typedef __attribute__((ext_vector_type(4)))  int i32x4;

union Frag {
    v4u  u[2];   // 2 x 16 bytes
    v16bf v;     // 16 bf16
};

#if defined(__HIP_DEVICE_COMPILE__) && __has_builtin(__builtin_amdgcn_tensor_load_to_lds)
#define USE_TDM 1
#else
#define USE_TDM 0
#endif

// fp32 -> bf16, round to nearest even
__device__ __forceinline__ unsigned short f2bf(float f) {
    unsigned int u = __float_as_uint(f);
    u += 0x7FFFu + ((u >> 16) & 1u);
    return (unsigned short)(u >> 16);
}

__device__ __forceinline__ float mishf(float x) {
    float sp = (x > 20.0f) ? x : log1pf(__expf(x));
    return x * tanhf(sp);
}

#if USE_TDM
// Issue one TDM 2D tile load: global (dim1 rows x 32 bf16) -> LDS, with LDS row
// padding 64B data + 16B pad (matches the 40-ushort LDS stride used below).
// D# layout per CDNA5 ISA ch.8 (group0 128b, group1 256b, groups2/3 zero).
__device__ __forceinline__ void tdm_load_tile(const unsigned short* gptr,
                                              unsigned int lds_addr,
                                              int dim0 /*row len avail*/,
                                              int dim1 /*rows avail*/,
                                              int ld   /*row stride, elems*/) {
    const unsigned long long ga = (unsigned long long)(size_t)gptr;
    u32x4 g0;
    g0[0] = 1u;                                           // count=1, user desc
    g0[1] = lds_addr;                                     // lds_addr [63:32]
    g0[2] = (unsigned int)(ga & 0xFFFFFFFFu);             // global_addr lo
    g0[3] = (unsigned int)((ga >> 32) & 0x01FFFFFFu)      // global_addr hi (57b)
          | 0x80000000u;                                  // type=2 ("image")
    const unsigned int td0 = (unsigned int)dim0;
    const unsigned int td1 = (unsigned int)dim1;
    i32x8 g1;
    g1[0] = (int)((1u << 16)      // data_size = 2 bytes (bf16)
                | (1u << 20)      // pad_enable
                | (3u << 22)      // pad_interval: every 16 DWORDs (=64B row)
                | (3u << 25));    // pad_amount: 4 DWORDs (=16B) -> 80B stride
    g1[1] = (int)((td0 & 0xFFFFu) << 16);                   // tensor_dim0 lo
    g1[2] = (int)((td0 >> 16) | ((td1 & 0xFFFFu) << 16));   // dim0 hi | dim1 lo
    g1[3] = (int)((td1 >> 16) | (32u << 16));               // dim1 hi | tile_dim0=32
    g1[4] = (int)(128u);                                    // tile_dim1=128, tile_dim2=0
    g1[5] = (int)(unsigned int)ld;                          // tensor_dim0_stride lo
    g1[6] = 0;                                              // stride hi | dim1_stride lo
    g1[7] = 0;
    const i32x4 z4 = (i32x4)(0);
#if __clang_major__ >= 23
    __builtin_amdgcn_tensor_load_to_lds(g0, g1, z4, z4, (i32x8)(0), 0);
#else
    __builtin_amdgcn_tensor_load_to_lds(g0, g1, z4, z4, 0);
#endif
}
#endif

// ---------- fused transpose + fp32->bf16: in [K,N] f32 -> out [N,K] bf16 ----------
__global__ __launch_bounds__(256)
void cvt_t_f32_bf16(const float* __restrict__ in, unsigned short* __restrict__ out,
                    int K, int N) {
    __shared__ float tile[32][33];
    const int k0 = blockIdx.y * 32;
    const int n0 = blockIdx.x * 32;
    const int tx = threadIdx.x & 31;
    const int ty = threadIdx.x >> 5;
#pragma unroll
    for (int i = 0; i < 32; i += 8) {
        const int k = k0 + ty + i, n = n0 + tx;
        tile[ty + i][tx] = (k < K && n < N) ? in[(size_t)k * N + n] : 0.f;
    }
    __syncthreads();
#pragma unroll
    for (int i = 0; i < 32; i += 8) {
        const int n = n0 + ty + i, k = k0 + tx;
        if (n < N && k < K) out[(size_t)n * K + k] = f2bf(tile[tx][ty + i]);
    }
}

// ---------- bf16 V reshape: [B*S, D] -> per-head column-major [B*H, HD, S] ----------
__global__ __launch_bounds__(256)
void transpose_v_bf16(const unsigned short* __restrict__ in,
                      unsigned short* __restrict__ out) {
    __shared__ unsigned short tile[32][33];
    const int s0 = blockIdx.y * 32;
    const int c0 = blockIdx.x * 32;
    const int tx = threadIdx.x & 31;
    const int ty = threadIdx.x >> 5;
#pragma unroll
    for (int i = 0; i < 32; i += 8)
        tile[ty + i][tx] = in[(size_t)(s0 + ty + i) * 1024 + c0 + tx];
    __syncthreads();
#pragma unroll
    for (int i = 0; i < 32; i += 8) {
        const int c   = c0 + ty + i;
        const int tok = s0 + tx;
        const int b = tok >> 10, s = tok & 1023;
        const int h = c >> 6,  hd = c & 63;
        out[((size_t)((b * 16 + h) * 64 + hd)) * 1024 + s] = tile[tx][ty + i];
    }
}

// ---------- LayerNorm: one block per row, D = 1024, 256 threads ----------
__global__ __launch_bounds__(256)
void ln_kernel(const float* __restrict__ x, const float* __restrict__ g,
               const float* __restrict__ b, float* __restrict__ yf,
               unsigned short* __restrict__ ybf, int D) {
    const int row = blockIdx.x;
    const int t   = threadIdx.x;
    const float* xr = x + (size_t)row * D;

    float lv[4];
    float s = 0.f, sq = 0.f;
#pragma unroll
    for (int i = 0; i < 4; ++i) {
        float v = xr[t + i * 256];
        lv[i] = v; s += v; sq += v * v;
    }
    __shared__ float rs[256], rq[256];
    rs[t] = s; rq[t] = sq;
    __syncthreads();
    for (int o = 128; o > 0; o >>= 1) {
        if (t < o) { rs[t] += rs[t + o]; rq[t] += rq[t + o]; }
        __syncthreads();
    }
    const float mean = rs[0] * (1.0f / 1024.0f);
    const float var  = rq[0] * (1.0f / 1024.0f) - mean * mean;
    const float inv  = rsqrtf(var + 1e-5f);
#pragma unroll
    for (int i = 0; i < 4; ++i) {
        const int c = t + i * 256;
        const float y = (lv[i] - mean) * inv * g[c] + b[c];
        if (yf)  yf[(size_t)row * D + c] = y;
        if (ybf) ybf[(size_t)row * D + c] = f2bf(y);
    }
}

// ---------- WMMA GEMM: C = act(alpha * A@Bt^T + bias + resid) ----------
// A : M x K bf16 row-major (lda); Bt: N x K bf16 row-major (ldb)
// Tile 128x128x32, double-buffered LDS staged by the Tensor Data Mover
// (wave 0 issues tensor_load_to_lds; all waves sync on barrier after
//  s_wait_tensorcnt). Falls back to per-thread b128 staging if TDM builtin
// is unavailable. 256 threads = 8 waves, wave = 32x64 (2x4 WMMA tiles).
__global__ __launch_bounds__(256)
void wmma_gemm_bf16(const unsigned short* __restrict__ A, int lda,
                    const unsigned short* __restrict__ Bt, int ldb,
                    float* __restrict__ Cf, unsigned short* __restrict__ Cbf, int ldc,
                    const float* __restrict__ bias,
                    const float* __restrict__ resid,
                    int M, int N, int K, float alpha, int act) {
    __shared__ unsigned short As[2][128 * 40];   // [buf][row][k], k-stride 40 (pad)
    __shared__ unsigned short Bs[2][128 * 40];   // [buf][col][k]

    const int tid  = threadIdx.x;
    const int lane = tid & 31;
    const int wave = tid >> 5;
    const int wm   = wave & 3;
    const int wn   = wave >> 2;
    const int m0   = blockIdx.y * 128;
    const int n0   = blockIdx.x * 128;

    v8f acc[2][4];
#pragma unroll
    for (int tm = 0; tm < 2; ++tm)
#pragma unroll
        for (int tn = 0; tn < 4; ++tn) acc[tm][tn] = (v8f)(0.0f);

    const int nk = K >> 5;

#if USE_TDM
    const unsigned asbase = (unsigned int)(size_t)&As[0][0];
    const unsigned bsbase = (unsigned int)(size_t)&Bs[0][0];
    const unsigned short* agbase = A  + (size_t)m0 * lda;
    const unsigned short* bgbase = Bt + (size_t)n0 * ldb;
    // prologue: DMA tile 0 into buffer 0
    if (wave == 0) {
        tdm_load_tile(agbase, asbase, K, M - m0, lda);
        tdm_load_tile(bgbase, bsbase, K, N - n0, ldb);
        __builtin_amdgcn_s_wait_tensorcnt(0);
    }
    __syncthreads();
#else
    const int lrow = tid >> 1;
    const int lseg = (tid & 1) * 16;
    const int gm = m0 + lrow;
    const int gn = n0 + lrow;
    const bool amask = (gm < M);
    const bool bmask = (gn < N);
    const unsigned short* aptr = A  + (size_t)gm * lda + lseg;
    const unsigned short* bptr = Bt + (size_t)gn * ldb + lseg;
    {
        v4u a0 = (v4u)(0u), a1 = (v4u)(0u), b0 = (v4u)(0u), b1 = (v4u)(0u);
        if (amask) { const v4u* p = (const v4u*)aptr; a0 = p[0]; a1 = p[1]; }
        if (bmask) { const v4u* p = (const v4u*)bptr; b0 = p[0]; b1 = p[1]; }
        *(v4u*)&As[0][lrow * 40 + lseg]     = a0;
        *(v4u*)&As[0][lrow * 40 + lseg + 8] = a1;
        *(v4u*)&Bs[0][lrow * 40 + lseg]     = b0;
        *(v4u*)&Bs[0][lrow * 40 + lseg + 8] = b1;
    }
    __syncthreads();
#endif

    const int rA  = lane & 15;
    const int kbA = (lane >> 4) * 8;    // A frag: K chunks {kb..kb+7, kb+16..kb+23}
    const int ksB = (lane >> 4) * 8;    // B frag: K chunks {kb..kb+7, kb+16..kb+23}

    for (int ki = 0; ki < nk; ++ki) {
        const int cur = ki & 1;
        const int nxt = cur ^ 1;
        const bool have = (ki + 1) < nk;

#if USE_TDM
        // issue next-tile DMA; it runs underneath the WMMAs below
        if (have && wave == 0) {
            const int ko = (ki + 1) << 5;
            tdm_load_tile(agbase + ko, asbase + (unsigned)nxt * 10240u,
                          K - ko, M - m0, lda);
            tdm_load_tile(bgbase + ko, bsbase + (unsigned)nxt * 10240u,
                          K - ko, N - n0, ldb);
        }
#else
        v4u na0 = (v4u)(0u), na1 = (v4u)(0u), nb0 = (v4u)(0u), nb1 = (v4u)(0u);
        if (have) {
            const int ko = (ki + 1) << 5;
            if (amask) { const v4u* p = (const v4u*)(aptr + ko); na0 = p[0]; na1 = p[1]; }
            if (bmask) { const v4u* p = (const v4u*)(bptr + ko); nb0 = p[0]; nb1 = p[1]; }
        }
#endif

        // ---- compute from current buffer
        Frag a[2], b[4];
#pragma unroll
        for (int tm = 0; tm < 2; ++tm) {
            const unsigned short* p = &As[cur][(wm * 32 + tm * 16 + rA) * 40 + kbA];
            a[tm].u[0] = *(const v4u*)p;
            a[tm].u[1] = *(const v4u*)(p + 16);
        }
#pragma unroll
        for (int tn = 0; tn < 4; ++tn) {
            const unsigned short* p = &Bs[cur][(wn * 64 + tn * 16 + rA) * 40 + ksB];
            b[tn].u[0] = *(const v4u*)p;
            b[tn].u[1] = *(const v4u*)(p + 16);
        }
#pragma unroll
        for (int tm = 0; tm < 2; ++tm)
#pragma unroll
            for (int tn = 0; tn < 4; ++tn)
                acc[tm][tn] = __builtin_amdgcn_wmma_f32_16x16x32_bf16(
                    false, a[tm].v, false, b[tn].v,
                    (short)0, acc[tm][tn], false, false);

#if USE_TDM
        if (have && wave == 0) __builtin_amdgcn_s_wait_tensorcnt(0);
#else
        if (have) {
            *(v4u*)&As[nxt][lrow * 40 + lseg]     = na0;
            *(v4u*)&As[nxt][lrow * 40 + lseg + 8] = na1;
            *(v4u*)&Bs[nxt][lrow * 40 + lseg]     = nb0;
            *(v4u*)&Bs[nxt][lrow * 40 + lseg + 8] = nb1;
        }
#endif
        __syncthreads();
    }

    // ---- epilogue: C/D layout, VGPR r -> M = r + 8*(lane>=16), N = lane%16
    const int cn = lane & 15;
    const int cr = (lane >> 4) * 8;
#pragma unroll
    for (int tm = 0; tm < 2; ++tm) {
#pragma unroll
        for (int tn = 0; tn < 4; ++tn) {
#pragma unroll
            for (int r = 0; r < 8; ++r) {
                const int m = m0 + wm * 32 + tm * 16 + cr + r;
                const int n = n0 + wn * 64 + tn * 16 + cn;
                if (m < M && n < N) {
                    float v = acc[tm][tn][r] * alpha;
                    if (bias)  v += bias[n];
                    if (resid) v += resid[(size_t)m * ldc + n];
                    if (act == 1) v = mishf(v);
                    const size_t idx = (size_t)m * ldc + n;
                    if (Cf)  Cf[idx]  = v;
                    if (Cbf) Cbf[idx] = f2bf(v);
                }
            }
        }
    }
}

// ---------- exact 1.5-entmax over one row of 1024 scores ----------
__global__ __launch_bounds__(1024)
void entmax15_kernel(const float* __restrict__ scores,
                     unsigned short* __restrict__ attn_bf) {
    const int n = 1024;
    const int row = blockIdx.x;
    const int t = threadIdx.x;
    const float* sr = scores + (size_t)row * n;

    __shared__ float zs[1024];
    __shared__ float ca[1024];
    __shared__ float cb[1024];

    const float v = sr[t];
    ca[t] = v;
    __syncthreads();
    for (int o = 512; o > 0; o >>= 1) {
        if (t < o) ca[t] = fmaxf(ca[t], ca[t + o]);
        __syncthreads();
    }
    const float zmax = ca[0];
    __syncthreads();

    const float z = (v - zmax) * 0.5f;
    zs[t] = z;
    __syncthreads();

    // bitonic sort, descending
    for (int k = 2; k <= n; k <<= 1) {
        for (int j = k >> 1; j > 0; j >>= 1) {
            const int ixj = t ^ j;
            if (ixj > t) {
                const float a = zs[t], b = zs[ixj];
                const bool descend = ((t & k) == 0);
                if (descend ? (a < b) : (a > b)) { zs[t] = b; zs[ixj] = a; }
            }
            __syncthreads();
        }
    }

    // inclusive scans of zs and zs^2
    const float zsv = zs[t];
    ca[t] = zsv;
    cb[t] = zsv * zsv;
    __syncthreads();
    for (int d = 1; d < n; d <<= 1) {
        const float a1 = ca[t], b1 = cb[t];
        const float a0 = (t >= d) ? ca[t - d] : 0.f;
        const float b0 = (t >= d) ? cb[t - d] : 0.f;
        __syncthreads();
        ca[t] = a1 + a0;
        cb[t] = b1 + b0;
        __syncthreads();
    }

    const float kk = (float)(t + 1);
    const float mean   = ca[t] / kk;
    const float meansq = cb[t] / kk;
    const float ss     = kk * (meansq - mean * mean);
    const float delta  = (1.0f - ss) / kk;
    const float tau    = mean - sqrtf(fmaxf(delta, 0.f));
    __syncthreads();
    ca[t] = tau;
    cb[t] = (tau <= zsv) ? 1.f : 0.f;
    __syncthreads();
    for (int o = 512; o > 0; o >>= 1) {
        if (t < o) cb[t] += cb[t + o];
        __syncthreads();
    }
    const int support = (int)cb[0];
    const float tau_star = ca[support - 1];

    const float p = fmaxf(z - tau_star, 0.f);
    attn_bf[(size_t)row * n + t] = f2bf(p * p);
}

// ---------- host orchestration ----------
extern "C" void kernel_launch(void* const* d_in, const int* in_sizes, int n_in,
                              void* d_out, int out_size, void* d_ws, size_t ws_size,
                              hipStream_t stream) {
    (void)in_sizes; (void)n_in; (void)out_size; (void)ws_size;
    constexpr int Bz = 4, S = 1024, D = 1024, H = 16, HD = 64, FF = 4096;
    constexpr int BS = Bz * S;
    const float SCALE = 0.125f;   // 1/sqrt(64)

    const float* x      = (const float*)d_in[0];
    const float* wq     = (const float*)d_in[1];
    const float* wk     = (const float*)d_in[2];
    const float* wv     = (const float*)d_in[3];
    const float* wo     = (const float*)d_in[4];
    const float* bq     = (const float*)d_in[5];
    const float* bk     = (const float*)d_in[6];
    const float* bv     = (const float*)d_in[7];
    const float* bo     = (const float*)d_in[8];
    const float* w_up   = (const float*)d_in[9];
    const float* b_up   = (const float*)d_in[10];
    const float* w_down = (const float*)d_in[11];
    const float* b_down = (const float*)d_in[12];
    const float* ln1_g  = (const float*)d_in[13];
    const float* ln1_b  = (const float*)d_in[14];
    const float* ln2_g  = (const float*)d_in[15];
    const float* ln2_b  = (const float*)d_in[16];
    const float* lnf_g  = (const float*)d_in[17];
    const float* lnf_b  = (const float*)d_in[18];
    float* out = (float*)d_out;

    char* ws = (char*)d_ws;
    size_t off = 0;
    auto alloc = [&](size_t bytes) -> void* {
        void* p = ws + off;
        off += (bytes + 255) & ~(size_t)255;
        return p;
    };
    unsigned short* ybf    = (unsigned short*)alloc((size_t)BS * D * 2);
    unsigned short* wqT    = (unsigned short*)alloc((size_t)D * D * 2);
    unsigned short* wkT    = (unsigned short*)alloc((size_t)D * D * 2);
    unsigned short* wvT    = (unsigned short*)alloc((size_t)D * D * 2);
    unsigned short* woT    = (unsigned short*)alloc((size_t)D * D * 2);
    unsigned short* wupT   = (unsigned short*)alloc((size_t)D * FF * 2);
    unsigned short* wdnT   = (unsigned short*)alloc((size_t)FF * D * 2);
    unsigned short* qbf    = (unsigned short*)alloc((size_t)BS * D * 2);
    unsigned short* kbf    = (unsigned short*)alloc((size_t)BS * D * 2);
    unsigned short* vbf    = (unsigned short*)alloc((size_t)BS * D * 2);
    unsigned short* vT     = (unsigned short*)alloc((size_t)BS * D * 2);
    float*          sc     = (float*)alloc((size_t)S * S * 4);
    unsigned short* attnbf = (unsigned short*)alloc((size_t)S * S * 2);
    unsigned short* obf    = (unsigned short*)alloc((size_t)BS * D * 2);
    float*          x1     = (float*)alloc((size_t)BS * D * 4);
    unsigned short* y2bf   = (unsigned short*)alloc((size_t)BS * D * 2);
    unsigned short* upbf   = (unsigned short*)alloc((size_t)BS * FF * 2);
    float*          x2     = (float*)alloc((size_t)BS * D * 4);

    auto cvtT = [&](const float* src, unsigned short* dst, int K, int N) {
        dim3 grid((N + 31) / 32, (K + 31) / 32, 1);
        cvt_t_f32_bf16<<<grid, 256, 0, stream>>>(src, dst, K, N);
    };
    auto gemm = [&](const unsigned short* A, int lda,
                    const unsigned short* Bt, int ldb,
                    float* Cf, unsigned short* Cbf, int ldc,
                    const float* bias, const float* resid,
                    int M, int N, int K, float alpha, int act) {
        dim3 grid((N + 127) / 128, (M + 127) / 128, 1);
        wmma_gemm_bf16<<<grid, 256, 0, stream>>>(A, lda, Bt, ldb,
                                                 Cf, Cbf, ldc, bias, resid,
                                                 M, N, K, alpha, act);
    };

    cvtT(wq, wqT, D, D);
    cvtT(wk, wkT, D, D);
    cvtT(wv, wvT, D, D);
    cvtT(wo, woT, D, D);
    cvtT(w_up, wupT, D, FF);
    cvtT(w_down, wdnT, FF, D);

    ln_kernel<<<BS, 256, 0, stream>>>(x, ln1_g, ln1_b, nullptr, ybf, D);

    gemm(ybf, D, wqT, D, nullptr, qbf, D, bq, nullptr, BS, D, D, 1.0f, 0);
    gemm(ybf, D, wkT, D, nullptr, kbf, D, bk, nullptr, BS, D, D, 1.0f, 0);
    gemm(ybf, D, wvT, D, nullptr, vbf, D, bv, nullptr, BS, D, D, 1.0f, 0);

    transpose_v_bf16<<<dim3(D / 32, BS / 32, 1), 256, 0, stream>>>(vbf, vT);

    for (int b = 0; b < Bz; ++b) {
        for (int h = 0; h < H; ++h) {
            const size_t hoff = (size_t)b * S * D + (size_t)h * HD;
            const size_t vto  = (size_t)(b * H + h) * HD * S;
            gemm(qbf + hoff, D, kbf + hoff, D,
                 sc, nullptr, S, nullptr, nullptr, S, S, HD, SCALE, 0);
            entmax15_kernel<<<S, 1024, 0, stream>>>(sc, attnbf);
            gemm(attnbf, S, vT + vto, S,
                 nullptr, obf + hoff, D, nullptr, nullptr, S, HD, S, 1.0f, 0);
        }
    }

    gemm(obf, D, woT, D, x1, nullptr, D, bo, x, BS, D, D, 1.0f, 0);
    ln_kernel<<<BS, 256, 0, stream>>>(x1, ln2_g, ln2_b, nullptr, y2bf, D);
    gemm(y2bf, D, wupT, D, nullptr, upbf, FF, b_up, nullptr, BS, FF, D, 1.0f, 1);
    gemm(upbf, FF, wdnT, FF, x2, nullptr, D, b_down, x1, BS, D, FF, 1.0f, 0);
    ln_kernel<<<BS, 256, 0, stream>>>(x2, lnf_g, lnf_b, out, nullptr, D);
}
